// MelSpectrogram_28174985462359
// MI455X (gfx1250) — compile-verified, matrix operand order checked by reading, not run on hardware
//
#include <hip/hip_runtime.h>
#include <hip/hip_bf16.h>

typedef __attribute__((ext_vector_type(16))) _Float16 v16h;
typedef __attribute__((ext_vector_type(8)))  _Float16 v8h;
typedef __attribute__((ext_vector_type(8)))  float    v8f;

// ---- problem constants (match reference) ----
constexpr int kBsz   = 32;
constexpr int kT     = 262144;
constexpr int kPad   = 384;                  // (1024-256)/2
constexpr int kPLen  = kT + 2 * kPad;        // 262912
constexpr int kHop   = 256;
constexpr int kNfft  = 1024;
constexpr int kFrames = 1024;                // (262912-1024)/256 + 1
constexpr int kCut   = 513;                  // N_FFT/2 + 1
constexpr int kCPad  = 544;                  // 17 * 32 : padded freq dim for WMMA K
constexpr int kBasisRows    = 1026;          // 2*kCut
constexpr int kBasisRowsPad = 1056;          // headroom so imag-row tiles never run off the end
constexpr int kMels  = 80;

// ---- workspace layout (all 16B aligned) ----
constexpr size_t kOffAudio = 0;                                             // f16 [32][262912]
constexpr size_t kOffBasis = kOffAudio + (size_t)kBsz * kPLen * 2;          // f16 [1056][1024]
constexpr size_t kOffMelB  = kOffBasis + (size_t)kBasisRowsPad * kNfft * 2; // f16 [80][544]
constexpr size_t kOffMagT  = kOffMelB + (size_t)kMels * kCPad * 2;          // f16 [32][1024][544]
// total ~54.7 MB

// ============ prep kernels ============

// reflect-pad audio and convert f32 -> f16. grid (1027, 32) x 256
__global__ void k_prep_audio(const float* __restrict__ audio, _Float16* __restrict__ out) {
    int p = blockIdx.x * 256 + threadIdx.x;            // 0..262911 (1027*256 == 262912)
    int b = blockIdx.y;
    int i = p - kPad;
    if (i < 0) i = -i;
    else if (i >= kT) i = 2 * kT - 2 - i;
    out[(size_t)b * kPLen + p] = (_Float16)audio[(size_t)b * kT + i];
}

// basis f32[1026][1024] -> f16[1056][1024], zero pad rows. grid 4224 x 256
__global__ void k_prep_basis(const float* __restrict__ basis, _Float16* __restrict__ out) {
    int idx = blockIdx.x * 256 + threadIdx.x;          // < 1056*1024
    int row = idx >> 10, col = idx & 1023;
    float v = (row < kBasisRows) ? basis[(size_t)row * kNfft + col] : 0.0f;
    out[idx] = (_Float16)v;
}

// mel basis f32[80][513] -> f16[80][544], zero pad cols. grid 170 x 256
__global__ void k_prep_mel(const float* __restrict__ mb, _Float16* __restrict__ out) {
    int idx = blockIdx.x * 256 + threadIdx.x;          // < 80*544
    int row = idx / kCPad, col = idx % kCPad;
    float v = (col < kCut) ? mb[(size_t)row * kCut + col] : 0.0f;
    out[idx] = (_Float16)v;
}

// ============ WMMA micro-helper ============
__device__ __forceinline__ v16h ld_frag(const _Float16* p) {
    // two contiguous 16B chunks at p and p+16 (A/B fragment per-lane pattern,
    // ISA 7.12.2: lane<16 holds K {0..7,16..23}, lane>=16 holds K {8..15,24..31})
    v8h lo = *(const v8h*)(p);
    v8h hi = *(const v8h*)(p + 16);
    return __builtin_shufflevector(lo, hi, 0,1,2,3,4,5,6,7,8,9,10,11,12,13,14,15);
}

#define WMMA_F16(A, B, C) \
    __builtin_amdgcn_wmma_f32_16x16x32_f16(false, (A), false, (B), (short)0, (C), false, false)

// ============ stage 1: STFT + magnitude (real+imag fused, 2 frame-subtiles/wave) ====
// grid (34 c-tiles, 4 frame-tiles of 256, 32 batch) x 256 threads (8 waves).
// Each wave owns TWO 16-frame subtiles and accumulates both the real tile
// (basis rows c0..c0+15) and the imag tile (rows 513+c0..) for each:
// 4 WMMAs per 4 fragment loads per k-step (A pair reused across both B's).
// Magnitude is formed in registers (acc elem r -> c = c0 + r + 8*lhalf,
// t = frame) and stored as contiguous b128 f16 stores into mag_t[b][t][c].
__global__ void __launch_bounds__(256) k_stft_mag(const _Float16* __restrict__ aud,
                                                  const _Float16* __restrict__ basis,
                                                  _Float16* __restrict__ magt) {
    const int c0 = blockIdx.x * 16;        // 0..528
    const int t0 = blockIdx.y * 256;       // frame tile base
    const int b  = blockIdx.z;
    _Float16* magb = magt + (size_t)b * kFrames * kCPad;

    if (c0 >= kCut) {                      // deterministically zero K-pad rows 528..543
        for (int i = threadIdx.x; i < 256 * 16; i += 256) {
            int t = i >> 4, c = i & 15;
            magb[(size_t)(t0 + t) * kCPad + (c0 + c)] = (_Float16)0.0f;
        }
        return;
    }

    const int wave  = threadIdx.x >> 5;
    const int lane  = threadIdx.x & 31;
    const int lhalf = lane >> 4;           // K-half selector per ISA A/B layout
    const int lmod  = lane & 15;

    const _Float16* arowR = basis + (size_t)(c0 + lmod) * kNfft + lhalf * 8;
    const _Float16* arowI = basis + (size_t)(kCut + c0 + lmod) * kNfft + lhalf * 8;
    const int frame0 = t0 + wave * 32 + lmod;              // subtile 0
    const _Float16* bcol = aud + (size_t)b * kPLen + (size_t)frame0 * kHop + lhalf * 8;
    // subtile 1 is frame0+16 -> constant element offset 16*kHop in the same stream

    v8f aR0 = {}, aI0 = {}, aR1 = {}, aI1 = {};
    #pragma unroll 4
    for (int k = 0; k < kNfft; k += 32) {
        v16h B0 = ld_frag(bcol + k);
        v16h B1 = ld_frag(bcol + k + 16 * kHop);
        v16h Ar = ld_frag(arowR + k);
        v16h Ai = ld_frag(arowI + k);
        aR0 = WMMA_F16(Ar, B0, aR0);
        aI0 = WMMA_F16(Ai, B0, aI0);
        aR1 = WMMA_F16(Ar, B1, aR1);
        aI1 = WMMA_F16(Ai, B1, aI1);
    }

    // magnitude in registers; cols >= 513 (c0==512 tile) are finite garbage * 0 mel weight
    v8h h0, h1;
    #pragma unroll
    for (int r = 0; r < 8; ++r) {
        h0[r] = (_Float16)__builtin_amdgcn_sqrtf(aR0[r] * aR0[r] + aI0[r] * aI0[r]);
        h1[r] = (_Float16)__builtin_amdgcn_sqrtf(aR1[r] * aR1[r] + aI1[r] * aI1[r]);
    }
    _Float16* dst0 = magb + (size_t)frame0 * kCPad + c0 + 8 * lhalf;
    *(v8h*)(dst0) = h0;
    *(v8h*)(dst0 + 16 * kCPad) = h1;       // frame0 + 16
}

// ============ stage 2: mel projection + log (2 frame-subtiles/wave) ============
// grid (5 m-tiles, 4 frame-tiles of 256, 32 batch) x 256 threads (8 waves).
__global__ void __launch_bounds__(256) k_mel(const _Float16* __restrict__ melb,   // [80][544]
                                             const _Float16* __restrict__ magt,   // [B][1024][544]
                                             float* __restrict__ out) {           // [B][80][1024]
    const int m0 = blockIdx.x * 16;
    const int t0 = blockIdx.y * 256;
    const int b  = blockIdx.z;
    const int wave  = threadIdx.x >> 5;
    const int lane  = threadIdx.x & 31;
    const int lhalf = lane >> 4;
    const int lmod  = lane & 15;

    const _Float16* arow = melb + (size_t)(m0 + lmod) * kCPad + lhalf * 8;
    const int t = t0 + wave * 32 + lmod;                   // subtile 0; subtile 1 at t+16
    const _Float16* bcol = magt + (size_t)b * kFrames * kCPad + (size_t)t * kCPad + lhalf * 8;

    v8f a0 = {}, a1 = {};
    #pragma unroll 4
    for (int k = 0; k < kCPad; k += 32) {
        v16h A  = ld_frag(arow + k);
        v16h B0 = ld_frag(bcol + k);
        v16h B1 = ld_frag(bcol + k + 16 * kCPad);
        a0 = WMMA_F16(A, B0, a0);
        a1 = WMMA_F16(A, B1, a1);
    }

    float* outb = out + (size_t)b * kMels * kFrames;
    #pragma unroll
    for (int r = 0; r < 8; ++r) {
        int M = r + 8 * lhalf;
        float v0 = a0[r]; v0 = (v0 < 1e-5f) ? 1e-5f : v0;
        float v1 = a1[r]; v1 = (v1 < 1e-5f) ? 1e-5f : v1;
        outb[(size_t)(m0 + M) * kFrames + t]      = __logf(v0);
        outb[(size_t)(m0 + M) * kFrames + t + 16] = __logf(v1);
    }
}

extern "C" void kernel_launch(void* const* d_in, const int* in_sizes, int n_in,
                              void* d_out, int out_size, void* d_ws, size_t ws_size,
                              hipStream_t stream) {
    const float* audio  = (const float*)d_in[0];   // [32][262144]
    const float* fbasis = (const float*)d_in[1];   // [1026][1024]
    const float* melb   = (const float*)d_in[2];   // [80][513]
    (void)d_in[3]; (void)in_sizes; (void)n_in; (void)out_size; (void)ws_size;

    char* ws = (char*)d_ws;
    _Float16* audio_h = (_Float16*)(ws + kOffAudio);
    _Float16* basis_h = (_Float16*)(ws + kOffBasis);
    _Float16* melb_h  = (_Float16*)(ws + kOffMelB);
    _Float16* magt    = (_Float16*)(ws + kOffMagT);

    dim3 blk(256);
    k_prep_audio<<<dim3(kPLen / 256, kBsz), blk, 0, stream>>>(audio, audio_h);
    k_prep_basis<<<dim3((kBasisRowsPad * kNfft) / 256), blk, 0, stream>>>(fbasis, basis_h);
    k_prep_mel  <<<dim3((kMels * kCPad) / 256), blk, 0, stream>>>(melb, melb_h);

    k_stft_mag<<<dim3(34, kFrames / 256, kBsz), blk, 0, stream>>>(audio_h, basis_h, magt);
    k_mel     <<<dim3(kMels / 16, kFrames / 256, kBsz), blk, 0, stream>>>(melb_h, magt, (float*)d_out);
}
